// Head_23149873725874
// MI455X (gfx1250) — matrix operand order repaired and checked by reading, not compiled, not run on hardware
//
#include <hip/hip_runtime.h>
#include <hip/hip_bf16.h>

typedef _Float16 h8   __attribute__((ext_vector_type(8)));
typedef _Float16 v16h __attribute__((ext_vector_type(16)));
typedef float    v8f  __attribute__((ext_vector_type(8)));
typedef float    f4   __attribute__((ext_vector_type(4)));
typedef unsigned u32x4 __attribute__((ext_vector_type(4)));
typedef int      i32x4 __attribute__((ext_vector_type(4)));
typedef int      i32x8 __attribute__((ext_vector_type(8)));

#define NB 32
#define NT 1024
#define NC 384
#define NH 64
#define NROWS (NB * NT)        // 32768
#define WSTRIDE 392            // padded LDS row stride (halfs) for W^T

#ifndef USE_TDM
#if defined(__has_builtin)
#if __has_builtin(__builtin_amdgcn_tensor_load_to_lds)
#define USE_TDM 1
#else
#define USE_TDM 0
#endif
#else
#define USE_TDM 0
#endif
#endif

static __device__ __forceinline__ v16h cat8(h8 lo, h8 hi) {
    return __builtin_shufflevector(lo, hi, 0,1,2,3,4,5,6,7,8,9,10,11,12,13,14,15);
}

// Low 32 bits of a flat pointer into LDS == byte offset in the workgroup's LDS.
static __device__ __forceinline__ unsigned lds_off(const void* p) {
    return (unsigned)(unsigned long long)(uintptr_t)p;
}

#if USE_TDM
// Tensor DMA descriptor per CDNA5 ISA.
// 2-D tile (tile_d1 rows of tile_d0 2-byte elements, row pitch stride0 elements);
// tile_d1 == 0 -> 1-D contiguous tile.  Huge tensor dims => no OOB clipping.
// clang-23 lane: 6-arg builtin (u32x4, i32x8, i32x4, i32x4, i32x8, i32 cpol);
// groups 2/3 + trailing group are zero (tensor is <= 2-D).
static __device__ __forceinline__ void tdm_load(unsigned lds, const void* gptr,
                                                unsigned tile_d0, unsigned tile_d1,
                                                unsigned long long stride0) {
    const unsigned long long ga = (unsigned long long)(uintptr_t)gptr;
    u32x4 g0;
    g0[0] = 1u;                                              // count=1, user mode
    g0[1] = lds;                                             // lds_addr
    g0[2] = (unsigned)ga;                                    // global_addr[31:0]
    g0[3] = (unsigned)((ga >> 32) & 0x01FFFFFFu) | (2u << 30); // addr[56:32] | type=2
    const unsigned td = 0x40000000u;                         // tensor_dim0/1 = huge
    i32x8 g1;
    g1[0] = (int)(1u << 16);                                 // data_size = 1 (2 bytes)
    g1[1] = (int)((td & 0xFFFFu) << 16);                     // tensor_dim0[15:0]
    g1[2] = (int)((td >> 16) | ((td & 0xFFFFu) << 16));      // td0[31:16] | td1[15:0]
    g1[3] = (int)((td >> 16) | (tile_d0 << 16));             // td1[31:16] | tile_dim0
    g1[4] = (int)tile_d1;                                    // tile_dim1 | tile_dim2=0
    g1[5] = (int)(unsigned)(stride0 & 0xFFFFFFFFull);        // dim0_stride[31:0]
    g1[6] = (int)(unsigned)(stride0 >> 32);                  // dim0_stride[47:32]
    g1[7] = 0;                                               // dim1_stride = 0
    i32x4 z4 = {0, 0, 0, 0};
    i32x8 z8 = {0, 0, 0, 0, 0, 0, 0, 0};
    __builtin_amdgcn_tensor_load_to_lds(g0, g1, z4, z4, z8, 0);
}
#endif

// ---------------------------------------------------------------------------
// Kernel 1: projections.  K = x@Wk, Q = 0.125 * x@Wq, V^T = (x@Wv)^T, all f16.
// Grid: 64 blocks x 256 threads.  2048 row-tiles / 512 waves = 4 tiles/wave.
// ---------------------------------------------------------------------------
__launch_bounds__(256)
__global__ void proj_kernel(const float* __restrict__ x,
                            const float* __restrict__ Wk,
                            const float* __restrict__ Wq,
                            const float* __restrict__ Wv,
                            _Float16* __restrict__ Qh,   // [NROWS][64], pre-scaled
                            _Float16* __restrict__ Kh,   // [NROWS][64]
                            _Float16* __restrict__ Vt)   // [NB][64][NT]
{
    __shared__ _Float16 Wt[NH * WSTRIDE];

    const int lane = threadIdx.x & 31;
    const int wid  = threadIdx.x >> 5;
    const int n    = lane & 15;
    const int hh   = lane >> 4;

    const float* Ws[3] = { Wk, Wq, Wv };
    const int wave_global = blockIdx.x * 8 + wid;

    for (int mat = 0; mat < 3; ++mat) {
        const float* W = Ws[mat];
        for (int idx = threadIdx.x; idx < NC * NH; idx += 256) {
            int k = idx >> 6, h = idx & 63;
            Wt[h * WSTRIDE + k] = (_Float16)W[idx];
        }
        __syncthreads();

        for (int tt = 0; tt < 4; ++tt) {
            const int R = (wave_global * 4 + tt) * 16;
            v8f acc[4] = {};
            const float* xrow = x + (size_t)(R + n) * NC;

            for (int kc = 0; kc < 12; ++kc) {
                f4 a0 = *(const f4*)(xrow + kc * 32 + 8 * hh);
                f4 a1 = *(const f4*)(xrow + kc * 32 + 8 * hh + 4);
                f4 a2 = *(const f4*)(xrow + kc * 32 + 16 + 8 * hh);
                f4 a3 = *(const f4*)(xrow + kc * 32 + 16 + 8 * hh + 4);
                v16h A;
#pragma unroll
                for (int i = 0; i < 4; ++i) {
                    A[i]      = (_Float16)a0[i];
                    A[4 + i]  = (_Float16)a1[i];
                    A[8 + i]  = (_Float16)a2[i];
                    A[12 + i] = (_Float16)a3[i];
                }
#pragma unroll
                for (int nt = 0; nt < 4; ++nt) {
                    const _Float16* wp = &Wt[(nt * 16 + n) * WSTRIDE + kc * 32 + 8 * hh];
                    acc[nt] = __builtin_amdgcn_wmma_f32_16x16x32_f16(
                        false, A, false, cat8(*(const h8*)wp, *(const h8*)(wp + 16)),
                        (short)0, acc[nt], false, false);
                }
            }

            if (mat < 2) {
                _Float16* dst = (mat == 0) ? Kh : Qh;
                const float s = (mat == 1) ? 0.125f : 1.0f;
#pragma unroll
                for (int nt = 0; nt < 4; ++nt)
#pragma unroll
                    for (int r = 0; r < 8; ++r)
                        dst[(size_t)(R + r + 8 * hh) * NH + nt * 16 + n] =
                            (_Float16)(acc[nt][r] * s);
            } else {
                const int b = R >> 10, t0 = R & 1023;
#pragma unroll
                for (int nt = 0; nt < 4; ++nt) {
                    h8 v;
#pragma unroll
                    for (int r = 0; r < 8; ++r) v[r] = (_Float16)acc[nt][r];
                    *(h8*)(Vt + ((size_t)b * NH + nt * 16 + n) * NT + t0 + 8 * hh) = v;
                }
            }
        }
        __syncthreads();
    }
}

// ---------------------------------------------------------------------------
// Kernel 2: causal flash attention.
// Block = 4 waves = 64 consecutive Q rows sharing TDM-staged K/V tiles (32 keys).
// Double-buffered: TDM for block j+1 overlaps compute of block j.
// ---------------------------------------------------------------------------
__launch_bounds__(128)
__global__ void attn_kernel(const _Float16* __restrict__ Qh,
                            const _Float16* __restrict__ Kh,
                            const _Float16* __restrict__ Vt,
                            float* __restrict__ out)
{
    __shared__ __align__(16) _Float16 Kbuf[2][32 * 64];  // [key][h]   8 KB
    __shared__ __align__(16) _Float16 Vbuf[2][64 * 32];  // [h][key]   8 KB
    __shared__ __align__(16) _Float16 Pbuf[4][16 * 48];  // per-wave P 6 KB

    const int lane = threadIdx.x & 31;
    const int wid  = threadIdx.x >> 5;
    const int n    = lane & 15;
    const int hh   = lane >> 4;
    _Float16* P = Pbuf[wid];

    const int b       = blockIdx.x >> 4;        // 16 WGs per batch
    const int wg_t    = blockIdx.x & 15;
    const int wg_row0 = wg_t * 64;
    const int row0    = wg_row0 + wid * 16;     // this wave's Q rows
    const int my_jmax = (row0 + 15) >> 5;
    const int Jmax    = 2 * wg_t + 1;           // == (wg_row0 + 63) >> 5

    const _Float16* Qb = Qh + (size_t)b * NT * NH;
    const _Float16* Kb = Kh + (size_t)b * NT * NH;
    const _Float16* Vb = Vt + (size_t)b * NH * NT;

    // Q tile as two A operands (h-chunks of 32)
    v16h AQ[2];
#pragma unroll
    for (int c = 0; c < 2; ++c) {
        const _Float16* qp = Qb + (size_t)(row0 + n) * NH + 32 * c + 8 * hh;
        AQ[c] = cat8(*(const h8*)(qp), *(const h8*)(qp + 16));
    }

    v16h ONES;
#pragma unroll
    for (int i = 0; i < 16; ++i) ONES[i] = (_Float16)1.0f;

    v8f o[4] = {};
    v8f lsum = {};
    float m_run = -1e30f;

#if USE_TDM
    if (wid == 0) {   // prologue: stage block 0
        tdm_load(lds_off(&Kbuf[0][0]), Kb, 2048, 0, 0);            // 1-D, 4 KB
        tdm_load(lds_off(&Vbuf[0][0]), Vb, 32, 64, (unsigned long long)NT);
    }
#endif

    for (int j = 0; j <= Jmax; ++j) {
        const int kb = j * 32;
        const int bsel = j & 1;
#if USE_TDM
        if (wid == 0) __builtin_amdgcn_s_wait_tensorcnt(0);
        __syncthreads();                       // buf[bsel] ready; prior reads done
        if (wid == 0 && j < Jmax) {            // overlap next DMA with compute
            const int kn = kb + 32;
            tdm_load(lds_off(&Kbuf[bsel ^ 1][0]), Kb + (size_t)kn * NH, 2048, 0, 0);
            tdm_load(lds_off(&Vbuf[bsel ^ 1][0]), Vb + kn, 32, 64,
                     (unsigned long long)NT);
        }
#else
        __syncthreads();                       // prior reads of buf done
        // fallback staging: plain loads -> LDS
        for (int i = threadIdx.x; i < 256; i += 128) {
            *(h8*)(&Kbuf[bsel][i * 8]) = *(const h8*)(Kb + (size_t)kb * NH + i * 8);
            const int row = i >> 2, off = (i & 3) * 8;
            *(h8*)(&Vbuf[bsel][row * 32 + off]) =
                *(const h8*)(Vb + (size_t)row * NT + kb + off);
        }
        __syncthreads();
#endif
        if (j > my_jmax) continue;             // wave-uniform; barriers stay matched

        const _Float16* KT = Kbuf[bsel];
        const _Float16* VT = Vbuf[bsel];

        // ---- S = Q @ K^T  (16 x 32 scores) ----
        v8f s0 = {}, s1 = {};
#pragma unroll
        for (int c = 0; c < 2; ++c) {
            const _Float16* kp0 = KT + (n) * NH + 32 * c + 8 * hh;
            s0 = __builtin_amdgcn_wmma_f32_16x16x32_f16(
                false, AQ[c], false, cat8(*(const h8*)kp0, *(const h8*)(kp0 + 16)),
                (short)0, s0, false, false);
            const _Float16* kp1 = KT + (16 + n) * NH + 32 * c + 8 * hh;
            s1 = __builtin_amdgcn_wmma_f32_16x16x32_f16(
                false, AQ[c], false, cat8(*(const h8*)kp1, *(const h8*)(kp1 + 16)),
                (short)0, s1, false, false);
        }

        float pv0[8], pv1[8];
#pragma unroll
        for (int r = 0; r < 8; ++r) { pv0[r] = s0[r]; pv1[r] = s1[r]; }

        if (j == my_jmax) {                    // causal mask, diagonal block only
#pragma unroll
            for (int r = 0; r < 8; ++r) {
                const int mg = row0 + r + 8 * hh;
                if (kb + n      > mg) pv0[r] = -1e30f;
                if (kb + 16 + n > mg) pv1[r] = -1e30f;
            }
        }

        // ---- tile-wide online max (5 shuffles) ----
        float mx = fmaxf(pv0[0], pv1[0]);
#pragma unroll
        for (int r = 1; r < 8; ++r) mx = fmaxf(mx, fmaxf(pv0[r], pv1[r]));
        mx = fmaxf(mx, __shfl_xor(mx, 1));
        mx = fmaxf(mx, __shfl_xor(mx, 2));
        mx = fmaxf(mx, __shfl_xor(mx, 4));
        mx = fmaxf(mx, __shfl_xor(mx, 8));
        mx = fmaxf(mx, __shfl_xor(mx, 16));
        const float mnew = fmaxf(m_run, mx);
        if (mnew > m_run) {                    // wave-uniform rescale
            const float alpha = __expf(m_run - mnew);
#pragma unroll
            for (int t = 0; t < 4; ++t)
#pragma unroll
                for (int r = 0; r < 8; ++r) o[t][r] *= alpha;
#pragma unroll
            for (int r = 0; r < 8; ++r) lsum[r] *= alpha;
        }
        m_run = mnew;

        // ---- P = exp(S - m), to per-wave LDS (C-layout -> A-layout) ----
#pragma unroll
        for (int r = 0; r < 8; ++r) {
            const int prow = (r + 8 * hh) * 48 + n;
            P[prow]      = (_Float16)__expf(pv0[r] - mnew);
            P[prow + 16] = (_Float16)__expf(pv1[r] - mnew);
        }
        asm volatile("s_wait_dscnt 0" ::: "memory");

        const _Float16* pp = P + n * 48 + 8 * hh;
        const v16h AP = cat8(*(const h8*)(pp), *(const h8*)(pp + 16));

        // ---- O += P @ V;  row sums via extra WMMA against all-ones B ----
#pragma unroll
        for (int nt = 0; nt < 4; ++nt) {
            const _Float16* vp = VT + (nt * 16 + n) * 32 + 8 * hh;
            o[nt] = __builtin_amdgcn_wmma_f32_16x16x32_f16(
                false, AP, false, cat8(*(const h8*)vp, *(const h8*)(vp + 16)),
                (short)0, o[nt], false, false);
        }
        lsum = __builtin_amdgcn_wmma_f32_16x16x32_f16(
            false, AP, false, ONES, (short)0, lsum, false, false);
    }

    // ---- epilogue ----
    float* ob = out + ((size_t)b * NT + row0) * NH;
#pragma unroll
    for (int r = 0; r < 8; ++r) {
        const float inv = 1.0f / lsum[r];
#pragma unroll
        for (int nt = 0; nt < 4; ++nt)
            ob[(size_t)(r + 8 * hh) * NH + nt * 16 + n] = o[nt][r] * inv;
    }
}

// ---------------------------------------------------------------------------
extern "C" void kernel_launch(void* const* d_in, const int* in_sizes, int n_in,
                              void* d_out, int out_size, void* d_ws, size_t ws_size,
                              hipStream_t stream) {
    const float* x  = (const float*)d_in[0];
    const float* Wk = (const float*)d_in[1];
    const float* Wq = (const float*)d_in[2];
    const float* Wv = (const float*)d_in[3];
    float* out = (float*)d_out;

    _Float16* Qh = (_Float16*)d_ws;
    _Float16* Kh = Qh + (size_t)NROWS * NH;
    _Float16* Vt = Kh + (size_t)NROWS * NH;

    proj_kernel<<<64, 256, 0, stream>>>(x, Wk, Wq, Wv, Qh, Kh, Vt);
    attn_kernel<<<512, 128, 0, stream>>>(Qh, Kh, Vt, out);
}
